// IFIB_50422916055156
// MI455X (gfx1250) — matrix operand order repaired and checked by reading, not compiled, not run on hardware
//
#include <hip/hip_runtime.h>
#include <hip/hip_bf16.h>
#include <cstdint>

// ============================================================================
// MI455X (gfx1250) implementation.
//  - rb conv stack  -> implicit GEMM on v_wmma_f32_16x16x32_f16 (f32->f16
//    convert on load, f32 accumulate); weight tiles staged to LDS by the
//    Tensor Data Mover (tensor_load_to_lds + s_wait_tensorcnt, verified to
//    lower in round 2).
//  - rfft2/irfft2   -> DFT-as-GEMM on v_wmma_f32_16x16x4_f32 (lowers).
//    320 = 2^6*5 is mixed-radix, so dense DFT matrices on the matrix engine
//    win; matrices + spectra stay in the 192MB L2. B tiles (shared by the 4
//    M-waves of a block) are TDM-DMA'd into LDS and read via volatile loads
//    (clang does not model the TDM builtin as writing LDS, so non-volatile
//    reads of the staged tile get folded away).
//  - wave32 everywhere; all tile sizes divide exactly so EXEC is all-ones
//    around every WMMA (ISA requirement).
// Param indexing assumes dict-insertion-order flattening of setup_inputs().
// ============================================================================

#define NPAD 320
#define NFRE 161
#define NFP  176          // 161 padded up to a multiple of 16
#define IMSZ (NPAD*NPAD)  // 102400
#define SPSZ (NPAD*NFP)   // 56320
#define IMH 256
#define IMW 256
#define HW  (IMH*IMW)
#define NB  2
#define CI  32
#define CO  64
#define NK  16
#define PHW 64            // psf tile 64x64
#define KC  16            // f32-GEMM K chunk staged in LDS (divides 320 & 176)

typedef float    v8f  __attribute__((ext_vector_type(8)));
typedef float    v2f  __attribute__((ext_vector_type(2)));
typedef _Float16 v16h __attribute__((ext_vector_type(16)));

#if defined(__has_builtin)
#  if __has_builtin(__builtin_amdgcn_wmma_f32_16x16x4_f32)
#    define HAVE_WMMA_F32 1
#  endif
#  if __has_builtin(__builtin_amdgcn_tensor_load_to_lds) && \
      __has_builtin(__builtin_amdgcn_s_wait_tensorcnt)
#    define HAVE_TDM 1
#  endif
#endif
#ifndef HAVE_WMMA_F32
#  define HAVE_WMMA_F32 0
#endif
#ifndef HAVE_TDM
#  define HAVE_TDM 0
#endif

// generic->LDS addressing: addr[31:0] is the LDS byte offset (ISA 10.2)
static __device__ __forceinline__ unsigned lds_offset_of(const void* p) {
  return (unsigned)(uintptr_t)p;
}

#if HAVE_TDM
typedef unsigned int v4u __attribute__((ext_vector_type(4)));
typedef int          v4i __attribute__((ext_vector_type(4)));
typedef int          v8i __attribute__((ext_vector_type(8)));

// Issue a TDM 2D tile load: rows x width_elems elements of size (1<<dsz) bytes
// from global (row stride in elements) into LDS at byte offset lds_off.
// D# packing per CDNA5 ISA 8.3/8.4 (group0: count/lds/global/type,
// group1: data_size, tensor dims, tile dims, dim0 stride).
static __device__ __forceinline__ void tdm_load_2d(
    const void* gsrc, unsigned lds_off, unsigned width_elems, unsigned rows,
    unsigned long long row_stride_elems, unsigned dsz_code) {
  unsigned long long ga = (unsigned long long)(uintptr_t)gsrc;
  v4u g0;
  g0[0] = 1u;                                   // count=1 (valid user D#)
  g0[1] = lds_off;                              // lds_addr (bytes)
  g0[2] = (unsigned)(ga & 0xFFFFFFFFu);         // global_addr[31:0]
  g0[3] = (unsigned)((ga >> 32) & 0x01FFFFFFu)  // global_addr[56:32]
          | (2u << 30);                         // type=2 ("image")
  v8i g1;
  g1[0] = (int)(dsz_code << 16);                               // data_size; wg_mask=0
  g1[1] = (int)((width_elems & 0xFFFFu) << 16);                // tensor_dim0[15:0]
  g1[2] = (int)((width_elems >> 16) | ((rows & 0xFFFFu) << 16)); // dim0 hi | dim1 lo
  g1[3] = (int)((rows >> 16) | ((width_elems & 0xFFFFu) << 16)); // dim1 hi | tile_dim0
  g1[4] = (int)(rows & 0xFFFFu);                               // tile_dim1 | tile_dim2=0
  g1[5] = (int)(row_stride_elems & 0xFFFFFFFFull);             // dim0_stride lo
  g1[6] = (int)((row_stride_elems >> 32) & 0xFFFFull);         // stride hi | stride1=0
  g1[7] = 0;
  v4i g2 = {0, 0, 0, 0}, g3 = {0, 0, 0, 0};
#if __clang_major__ >= 23
  v8i g4 = {0, 0, 0, 0, 0, 0, 0, 0};
  __builtin_amdgcn_tensor_load_to_lds(g0, g1, g2, g3, g4, 0);
#else
  __builtin_amdgcn_tensor_load_to_lds(g0, g1, g2, g3, 0);
#endif
}
#endif // HAVE_TDM

static __device__ __forceinline__ float geluf(float v) {
  return 0.5f * v * (1.0f + erff(v * 0.7071067811865475f));
}

// ---------------------------------------------------------------------------
// small utility kernels
// ---------------------------------------------------------------------------
__global__ void k_fill(float* p, long long n, float v) {
  long long i = (long long)blockIdx.x * blockDim.x + threadIdx.x;
  if (i < n) p[i] = v;
}

__global__ void k_w2h(const float* __restrict__ w, _Float16* __restrict__ o, int n) {
  int i = blockIdx.x * blockDim.x + threadIdx.x;
  if (i < n) o[i] = (_Float16)w[i];
}

// DFT matrices: Cn/Sn (320x176, cols>=161 zero), Cf/Sf (320x320),
// IRr/IRi (176x320) = weighted inverse-rfft synthesis incl. 1/(320*320).
__global__ void k_init_dft(float* Cn, float* Sn, float* Cf, float* Sf,
                           float* IRr, float* IRi) {
  int i = blockIdx.x * blockDim.x + threadIdx.x;
  if (i >= IMSZ) return;
  int n = i / NPAD, j = i % NPAD;
  int t = (n * j) % NPAD;                       // exact angle reduction
  float ang = 6.283185307179586f * (float)t / (float)NPAD;
  float c = cosf(ang), s = sinf(ang);
  Cf[i] = c; Sf[i] = s;
  if (j < NFP) {
    int o = n * NFP + j; bool v = (j < NFRE);
    Cn[o] = v ? c : 0.0f; Sn[o] = v ? s : 0.0f;
  }
  if (n < NFP) {
    float w = (n < NFRE) ? ((n == 0 || n == NFRE - 1) ? 1.0f : 2.0f) : 0.0f;
    float sc = w / (float)IMSZ;
    int o = n * NPAD + j;
    IRr[o] =  sc * c;
    IRi[o] = -sc * s;
  }
}

// pad 256 -> 320 (32 each side); edge=1 -> replicate, else zero
__global__ void k_pad(const float* __restrict__ src, float* __restrict__ dst,
                      int nimg, int edge) {
  long long i = (long long)blockIdx.x * blockDim.x + threadIdx.x;
  if (i >= (long long)nimg * IMSZ) return;
  int img = (int)(i / IMSZ), p = (int)(i % IMSZ);
  int y = p / NPAD - 32, x = p % NPAD - 32;
  float v;
  if (edge) {
    int yc = min(max(y, 0), IMH - 1), xc = min(max(x, 0), IMW - 1);
    v = src[(long long)img * HW + yc * IMW + xc];
  } else {
    v = (y >= 0 && y < IMH && x >= 0 && x < IMW)
          ? src[(long long)img * HW + y * IMW + x] : 0.0f;
  }
  dst[i] = v;
}

// FWHM=2 gaussian 7-tap: w_t = 2^{-t^2} normalized
__global__ void k_gauss(const float* __restrict__ src, float* __restrict__ dst,
                        int nimg, int vert) {
  const float gw[7] = {0.0009174312f, 0.0293577982f, 0.2348623853f,
                       0.4697247706f, 0.2348623853f, 0.0293577982f, 0.0009174312f};
  long long i = (long long)blockIdx.x * blockDim.x + threadIdx.x;
  if (i >= (long long)nimg * IMSZ) return;
  int img = (int)(i / IMSZ), p = (int)(i % IMSZ);
  int y = p / NPAD, x = p % NPAD;
  const float* b = src + (long long)img * IMSZ;
  float a = 0.0f;
#pragma unroll
  for (int t = -3; t <= 3; ++t) {
    int yy = vert ? y + t : y, xx = vert ? x : x + t;
    if (yy >= 0 && yy < NPAD && xx >= 0 && xx < NPAD) a += gw[t + 3] * b[yy * NPAD + xx];
  }
  dst[i] = a;
}

__global__ void k_red_sum(const float* __restrict__ x, int L, float* out, float scale) {
  __shared__ float sm[256];
  const float* p = x + (long long)blockIdx.x * L;
  float a = 0.0f;
  for (int i = threadIdx.x; i < L; i += 256) a += p[i];
  sm[threadIdx.x] = a; __syncthreads();
  for (int o = 128; o; o >>= 1) {
    if ((int)threadIdx.x < o) sm[threadIdx.x] += sm[threadIdx.x + o];
    __syncthreads();
  }
  if (!threadIdx.x) out[blockIdx.x] = sm[0] * scale;
}

__global__ void k_psf_pad_iso(const float* __restrict__ psf, const float* sums,
                              const float* alpha, float* __restrict__ out) {
  long long i = (long long)blockIdx.x * blockDim.x + threadIdx.x;
  if (i >= (long long)NK * IMSZ) return;
  int k = (int)(i / IMSZ), p = (int)(i % IMSZ);
  int y = p / NPAD, x = p % NPAD;
  float v = 0.0f;
  if (y >= 128 && y < 192 && x >= 128 && x < 192)
    v = psf[k * PHW * PHW + (y - 128) * PHW + (x - 128)] /
        (fabsf(sums[k]) * alpha[k] + 1e-12f);
  out[i] = v;
}

__global__ void k_psf_mean(const float* __restrict__ psf, float* __restrict__ pm) {
  int p = blockIdx.x * blockDim.x + threadIdx.x;
  if (p >= PHW * PHW) return;
  float a = 0.0f;
  for (int k = 0; k < NK; ++k) a += psf[k * PHW * PHW + p];
  pm[p] = a / (float)NK;
}

__global__ void k_psf_pad_fso(const float* __restrict__ pm, const float* s,
                              const float* alpha, float* __restrict__ out) {
  int i = blockIdx.x * blockDim.x + threadIdx.x;
  if (i >= IMSZ) return;
  int y = i / NPAD, x = i % NPAD;
  float v = 0.0f;
  if (y >= 128 && y < 192 && x >= 128 && x < 192)
    v = pm[(y - 128) * PHW + (x - 128)] / (fabsf(s[0] * alpha[0]) + 1e-12f);
  out[i] = v;
}

// inv = M * conj(P) / (|P|^2 + eps + relu(w_c)), per (k fixed) over 64 images
__global__ void k_wiener(const float* __restrict__ Gr, const float* __restrict__ Gi,
                         const float* __restrict__ Pr, const float* __restrict__ Pi,
                         const float* __restrict__ wreg,
                         float* __restrict__ Wr, float* __restrict__ Wi) {
  long long i = (long long)blockIdx.x * blockDim.x + threadIdx.x;
  if (i >= 64LL * SPSZ) return;
  int img = (int)(i / SPSZ), p = (int)(i % SPSZ), c = img % CI;
  float pr = Pr[p], pi = Pi[p];
  float den = pr * pr + pi * pi + 1e-6f + fmaxf(wreg[c], 0.0f);
  float mr = Gr[i], mi = Gi[i];
  Wr[i] = (mr * pr + mi * pi) / den;
  Wi[i] = (mi * pr - mr * pi) / den;
}

__global__ void k_cmul(const float* __restrict__ Gr, const float* __restrict__ Gi,
                       const float* __restrict__ Pr, const float* __restrict__ Pi,
                       float* __restrict__ Vr, float* __restrict__ Vi) {
  long long i = (long long)blockIdx.x * blockDim.x + threadIdx.x;
  if (i >= 64LL * SPSZ) return;
  int p = (int)(i % SPSZ);
  float gr = Gr[i], gi = Gi[i], pr = Pr[p], pi = Pi[p];
  Vr[i] = gr * pr - gi * pi;
  Vi[i] = gr * pi + gi * pr;
}

// ifftshift (shift 160) + crop[32:288] (+ optional *roi, + optional accumulate)
__global__ void k_shift_crop(const float* __restrict__ Sp, const float* __restrict__ roi,
                             float* __restrict__ out, int acc) {
  long long i = (long long)blockIdx.x * blockDim.x + threadIdx.x;
  if (i >= 64LL * HW) return;
  int img = (int)(i / HW), p = (int)(i % HW);
  int y = p / IMW, x = p % IMW;
  int sy = (y + 192) % NPAD, sx = (x + 192) % NPAD;
  float v = Sp[(long long)img * IMSZ + sy * NPAD + sx];
  if (roi) v *= roi[p];
  out[i] = acc ? out[i] + v : v;
}

__global__ void k_gn_stats(const float* __restrict__ x, long long n, float* st) {
  __shared__ float s1[256], s2[256];
  float a = 0.0f, b = 0.0f;
  for (long long i = (long long)blockIdx.x * 256 + threadIdx.x; i < n;
       i += (long long)gridDim.x * 256) {
    float v = x[i]; a += v; b += v * v;
  }
  s1[threadIdx.x] = a; s2[threadIdx.x] = b; __syncthreads();
  for (int o = 128; o; o >>= 1) {
    if ((int)threadIdx.x < o) { s1[threadIdx.x] += s1[threadIdx.x + o];
                                s2[threadIdx.x] += s2[threadIdx.x + o]; }
    __syncthreads();
  }
  if (!threadIdx.x) { atomicAdd(&st[0], s1[0]); atomicAdd(&st[1], s2[0]); }
}

__global__ void k_gn_apply(const float* __restrict__ x, float* __restrict__ y,
                           const float* __restrict__ g, const float* __restrict__ bt,
                           const float* __restrict__ st, long long n, int C, int gelu) {
  long long i = (long long)blockIdx.x * blockDim.x + threadIdx.x;
  if (i >= n) return;
  float mu = st[0] / (float)n;
  float var = st[1] / (float)n - mu * mu;
  float inv = rsqrtf(var + 1e-5f);
  int c = (int)((i / HW) % C);
  float v = (x[i] - mu) * inv * g[c] + bt[c];
  if (gelu) v = geluf(v);
  y[i] = v;
}

__global__ void k_dw3(const float* __restrict__ x, const float* __restrict__ w,
                      const float* __restrict__ b, float* __restrict__ y) {
  long long i = (long long)blockIdx.x * blockDim.x + threadIdx.x;
  if (i >= (long long)NB * CO * HW) return;
  int p = (int)(i % HW); int ch = (int)(i / HW); int c = ch % CO;
  int py = p / IMW, px = p % IMW;
  const float* xb = x + (long long)ch * HW;
  float a = b[c];
#pragma unroll
  for (int t = 0; t < 9; ++t) {
    int yy = py + t / 3 - 1, xx = px + t % 3 - 1;
    if (yy >= 0 && yy < IMH && xx >= 0 && xx < IMW) a += w[c * 9 + t] * xb[yy * IMW + xx];
  }
  y[i] = a;
}

__global__ void k_gate(const float* __restrict__ z, float* __restrict__ o) {
  long long i = (long long)blockIdx.x * blockDim.x + threadIdx.x;
  if (i >= (long long)NB * CO * HW) return;
  long long bidx = i / ((long long)CO * HW), r = i % ((long long)CO * HW);
  const float* zb = z + bidx * 2LL * CO * HW;
  o[i] = zb[r] * zb[(long long)CO * HW + r];
}

__global__ void k_chan_mean(const float* __restrict__ x, float* __restrict__ cm) {
  __shared__ float sm[256];
  const float* p = x + (long long)blockIdx.x * HW;
  float a = 0.0f;
  for (int i = threadIdx.x; i < HW; i += 256) a += p[i];
  sm[threadIdx.x] = a; __syncthreads();
  for (int o = 128; o; o >>= 1) {
    if ((int)threadIdx.x < o) sm[threadIdx.x] += sm[threadIdx.x + o];
    __syncthreads();
  }
  if (!threadIdx.x) cm[blockIdx.x] = sm[0] / (float)HW;
}

__global__ void k_se(const float* __restrict__ cm, const float* __restrict__ w,
                     const float* __restrict__ b, float* __restrict__ sev) {
  int t = threadIdx.x;
  if (t >= NB * CO) return;
  int bi = t / CO, co = t % CO;
  float a = b[co];
  for (int ci = 0; ci < CO; ++ci) a += w[co * CO + ci] * cm[bi * CO + ci];
  sev[t] = a;
}

__global__ void k_scale_se(float* __restrict__ y, const float* __restrict__ sev) {
  long long i = (long long)blockIdx.x * blockDim.x + threadIdx.x;
  if (i >= (long long)NB * CO * HW) return;
  y[i] *= sev[i / HW];
}

__global__ void k_axpy_chan(float* __restrict__ out, const float* __restrict__ a,
                            const float* __restrict__ s, const float* __restrict__ b2,
                            int C, long long n) {
  long long i = (long long)blockIdx.x * blockDim.x + threadIdx.x;
  if (i >= n) return;
  out[i] = a[i] + s[(i / HW) % C] * b2[i];
}

__global__ void k_mix(float* __restrict__ out, const float* __restrict__ a,
                      const float* __restrict__ sa, const float* __restrict__ b2,
                      const float* __restrict__ sb, long long n) {
  long long i = (long long)blockIdx.x * blockDim.x + threadIdx.x;
  if (i >= n) return;
  out[i] = a[i] * sa[0] + b2[i] * sb[0];
}

// ---------------------------------------------------------------------------
// f32 GEMM: block = 4 waves stacked on M (64 rows) x one 16-col N tile.
// The shared B tile (KC x 16 f32) is DMA'd to LDS by the Tensor Data Mover
// (falls back to a cooperative copy if the builtin is absent), then each wave
// runs v_wmma_f32_16x16x4_f32 with A from global (L2-resident) and B from LDS
// via volatile ds loads. C = alpha*A@B + beta*C, batched via gridDim.z.
// ---------------------------------------------------------------------------
__global__ __launch_bounds__(128)
void k_gemm_f32(const float* __restrict__ A, const float* __restrict__ B,
                float* __restrict__ C, int M, int N, int K,
                int lda, int ldb, int ldc,
                long long sA, long long sB, long long sC,
                float alpha, float beta) {
  __shared__ float Bs[KC * 16];
  A += (long long)blockIdx.z * sA;
  B += (long long)blockIdx.z * sB;
  C += (long long)blockIdx.z * sC;
  const int wid  = threadIdx.x >> 5;
  const int lane = threadIdx.x & 31;
  const int hl = lane >> 4, l16 = lane & 15;
  const int mBase = blockIdx.y * 64 + 16 * wid;
  const int nBase = blockIdx.x * 16;
  volatile const float* Bv = Bs;     // TDM writes are invisible to the compiler

  v8f acc = {0.f, 0.f, 0.f, 0.f, 0.f, 0.f, 0.f, 0.f};
  if (beta != 0.0f) {
    float bs = beta / alpha;
#pragma unroll
    for (int r = 0; r < 8; ++r)
      acc[r] = bs * C[(mBase + r + 8 * hl) * ldc + nBase + l16];
  }
  const float* Arow = A + (mBase + l16) * lda;

  for (int k0 = 0; k0 < K; k0 += KC) {
    if (wid == 0) {
#if HAVE_TDM
      tdm_load_2d(B + k0 * ldb + nBase, lds_offset_of(Bs),
                  /*width=*/16u, /*rows=*/(unsigned)KC,
                  (unsigned long long)ldb, /*dsz=4B*/2u);
      __builtin_amdgcn_s_wait_tensorcnt(0);
      asm volatile("" ::: "memory");
#else
      for (int t = lane; t < KC * 16; t += 32)
        Bs[t] = B[(k0 + (t >> 4)) * ldb + nBase + (t & 15)];
#endif
    }
    __syncthreads();
    __builtin_prefetch(Arow + k0 + KC, 0, 3);   // global_prefetch_b8 next chunk
#if HAVE_WMMA_F32
#pragma unroll
    for (int kk = 0; kk < KC; kk += 4) {
      v2f a, b;
      a[0] = Arow[k0 + kk + 2 * hl + 0];
      a[1] = Arow[k0 + kk + 2 * hl + 1];
      b[0] = Bv[(kk + 2 * hl + 0) * 16 + l16];
      b[1] = Bv[(kk + 2 * hl + 1) * 16 + l16];
      acc = __builtin_amdgcn_wmma_f32_16x16x4_f32(false, a, false, b,
                                                  (short)0, acc, false, false);
    }
#else
    for (int kk = 0; kk < KC; ++kk) {
      float bv = Bv[kk * 16 + l16];
#pragma unroll
      for (int r = 0; r < 8; ++r)
        acc[r] += A[(mBase + r + 8 * hl) * lda + k0 + kk] * bv;
    }
#endif
    __syncthreads();
  }
#pragma unroll
  for (int r = 0; r < 8; ++r)
    C[(mBase + r + 8 * hl) * ldc + nBase + l16] = alpha * acc[r];
}

// ---------------------------------------------------------------------------
// implicit-GEMM conv (1x1 or 3x3 SAME), f16 operands, f32 accumulate.
// Block = 8 waves: 64 couts x 32 pixels. K = Cin*kh*kh (multiple of 32).
// Weight tile (64 x 32 f16 = 4KB) is TDM-DMA'd to LDS each K chunk and shared
// by all 8 waves (volatile ds reads); B (im2col) built in-register from global.
// ---------------------------------------------------------------------------
__global__ __launch_bounds__(256)
void k_conv_wmma(const float* __restrict__ X, const _Float16* __restrict__ Wh,
                 const float* __restrict__ bias, float* __restrict__ Y,
                 int Cin, int Cout, int kh) {
  __shared__ _Float16 As[64 * 32];
  const int wid = threadIdx.x >> 5;
  const int lane = threadIdx.x & 31;
  const int hl = lane >> 4, l16 = lane & 15;
  const int mBase = blockIdx.y * 64 + 16 * (wid & 3);
  const int nBase = blockIdx.x * 32 + 16 * (wid >> 2);
  const int Ktot = Cin * kh * kh;
  const int mLoc = 16 * (wid & 3) + l16;        // row within the LDS tile
  volatile const _Float16* Av = As;             // see k_gemm_f32 comment

  const int n = nBase + l16;
  const int bidx = n / HW;
  const int hw = n % HW;
  const int py = hw / IMW, px = hw % IMW;
  const float* Xb = X + (long long)bidx * Cin * HW;

  v8f acc = {0.f, 0.f, 0.f, 0.f, 0.f, 0.f, 0.f, 0.f};

  for (int k0 = 0; k0 < Ktot; k0 += 32) {
    if (wid == 0) {
#if HAVE_TDM
      tdm_load_2d(Wh + (long long)blockIdx.y * 64 * Ktot + k0, lds_offset_of(As),
                  /*width=*/32u, /*rows=*/64u,
                  (unsigned long long)Ktot, /*dsz=2B*/1u);
      __builtin_amdgcn_s_wait_tensorcnt(0);
      asm volatile("" ::: "memory");
#else
      for (int t = lane; t < 64 * 32; t += 32)
        As[t] = Wh[(long long)(blockIdx.y * 64 + (t >> 5)) * Ktot + k0 + (t & 31)];
#endif
    }
    __syncthreads();
    v16h a, b;
#pragma unroll
    for (int j = 0; j < 16; ++j) {
      // A (16x32 f16): lane<16 K = j|j+8 pattern; lane>=16 shifted by 8
      int ka = ((j < 8) ? j : j + 8) + 8 * hl;
      a[j] = Av[mLoc * 32 + ka];
      // B (32x16 f16): same K pattern as A
      int kb = k0 + ((j < 8) ? j : j + 8) + 8 * hl;
      float v;
      if (kh == 1) {
        v = Xb[kb * HW + hw];
      } else {
        int c = kb / 9, t = kb - c * 9;
        int yy = py + t / 3 - 1, xx = px + t % 3 - 1;
        v = (yy >= 0 && yy < IMH && xx >= 0 && xx < IMW)
              ? Xb[(long long)c * HW + yy * IMW + xx] : 0.0f;
      }
      b[j] = (_Float16)v;
    }
    acc = __builtin_amdgcn_wmma_f32_16x16x32_f16(false, a, false, b,
                                                 (short)0, acc, false, false);
    __syncthreads();
  }
#pragma unroll
  for (int r = 0; r < 8; ++r) {
    int m = mBase + r + 8 * hl;
    Y[((long long)bidx * Cout + m) * HW + hw] = acc[r] + bias[m];
  }
}

// ---------------------------------------------------------------------------
extern "C" void kernel_launch(void* const* d_in, const int* in_sizes, int n_in,
                              void* d_out, int out_size, void* d_ws, size_t ws_size,
                              hipStream_t stream) {
  (void)in_sizes; (void)n_in; (void)out_size; (void)ws_size;

  const float* invF = (const float*)d_in[0];
  const float* fwdF = (const float*)d_in[1];
  const float* psf  = (const float*)d_in[2];
  const float* roi  = (const float*)d_in[3];
  // params, dict-insertion order
  const float* fsoAlpha = (const float*)d_in[4];
  const float* fsoGnG   = (const float*)d_in[5];
  const float* fsoGnB   = (const float*)d_in[6];
  const float* isoAlpha = (const float*)d_in[7];
  const float* isoPsfW  = (const float*)d_in[8];
  const float* isoGnG   = (const float*)d_in[9];
  const float* isoGnB   = (const float*)d_in[10];
  const float* invMixS  = (const float*)d_in[11];
  const float* invMixC  = (const float*)d_in[12];
  const float* fwdMixS  = (const float*)d_in[13];
  const float* fwdMixC  = (const float*)d_in[14];

  struct RBP {
    const float *conv1_w, *conv1_b, *norm1_g, *norm1_b, *rn1_g, *rn1_b,
      *pw1_w, *pw1_b, *dw_w, *dw_b, *pw2_w, *pw2_b, *se_w, *se_b,
      *c3_w, *c3_b, *beta, *rn2_g, *rn2_b, *pw3_w, *pw3_b, *pw4_w, *pw4_b,
      *gamma, *conv2_w, *conv2_b, *norm2_g, *norm2_b;
  };
  auto loadRB = [&](int base) {
    RBP r; const float** pp = (const float**)&r;
    for (int i = 0; i < 28; ++i) pp[i] = (const float*)d_in[base + i];
    return r;
  };
  RBP rbI = loadRB(15), rbF = loadRB(43);

  // ---- workspace bump allocator (floats) ----
  float* WS = (float*)d_ws;
  size_t off = 0;
  auto alloc = [&](size_t nf) { float* p = WS + off; off += (nf + 63) & ~(size_t)63; return p; };
  float *Cn = alloc(SPSZ), *Sn = alloc(SPSZ), *Cf = alloc(IMSZ), *Sf = alloc(IMSZ),
        *IRr = alloc(SPSZ), *IRi = alloc(SPSZ);
  float *padA = alloc(64LL * IMSZ), *padB = alloc(64LL * IMSZ);
  float *Fr = alloc(64LL * SPSZ), *Fi = alloc(64LL * SPSZ);
  float *Gr = alloc(64LL * SPSZ), *Gi = alloc(64LL * SPSZ);
  float *Hr = alloc(64LL * SPSZ), *Hi = alloc(64LL * SPSZ);
  float *psfPad = alloc((size_t)NK * IMSZ);
  float *PFr = alloc((size_t)NK * SPSZ), *PFi = alloc((size_t)NK * SPSZ);
  float *PGr = alloc((size_t)NK * SPSZ), *PGi = alloc((size_t)NK * SPSZ);
  float *mixed = alloc((size_t)NB * CI * HW);
  float *invX = alloc((size_t)NB * CI * HW), *fwdX = alloc((size_t)NB * CI * HW);
  float *inA = alloc((size_t)NB * CI * HW), *inB = alloc((size_t)NB * CI * HW);
  float *xb = alloc((size_t)NB * CO * HW), *t64a = alloc((size_t)NB * CO * HW),
        *t64b = alloc((size_t)NB * CO * HW), *t64c = alloc((size_t)NB * CO * HW);
  float *t128 = alloc((size_t)NB * 2 * CO * HW);
  float *stats = alloc(8), *psfsum = alloc(NK), *pm = alloc(PHW * PHW),
        *ps = alloc(8), *cm = alloc(NB * CO), *sev = alloc(NB * CO);
  _Float16* wh = (_Float16*)alloc(64 * 576 / 2 + 64);

  auto GB = [](long long n) { return dim3((unsigned)((n + 255) / 256)); };

  auto gemm = [&](const float* A, const float* B, float* C, int M, int N, int Kd,
                  int lda, int ldb, int ldc, long long sA, long long sB, long long sC,
                  float alpha, float beta, int batch) {
    dim3 g(N / 16, M / 64, batch);
    k_gemm_f32<<<g, 128, 0, stream>>>(A, B, C, M, N, Kd, lda, ldb, ldc, sA, sB, sC, alpha, beta);
  };
  // rfft2: row DFT (real->complex, 161 padded to 176) then col DFT (complex)
  auto rfft2 = [&](const float* img, float* fr, float* fi, float* gr, float* gi, int batch) {
    gemm(img, Cn, fr, NPAD, NFP, NPAD, NPAD, NFP, NFP, IMSZ, 0, SPSZ,  1.f, 0.f, batch);
    gemm(img, Sn, fi, NPAD, NFP, NPAD, NPAD, NFP, NFP, IMSZ, 0, SPSZ, -1.f, 0.f, batch);
    gemm(Cf, fr, gr, NPAD, NFP, NPAD, NPAD, NFP, NFP, 0, SPSZ, SPSZ,  1.f, 0.f, batch);
    gemm(Sf, fi, gr, NPAD, NFP, NPAD, NPAD, NFP, NFP, 0, SPSZ, SPSZ,  1.f, 1.f, batch);
    gemm(Cf, fi, gi, NPAD, NFP, NPAD, NPAD, NFP, NFP, 0, SPSZ, SPSZ,  1.f, 0.f, batch);
    gemm(Sf, fr, gi, NPAD, NFP, NPAD, NPAD, NFP, NFP, 0, SPSZ, SPSZ, -1.f, 1.f, batch);
  };
  // irfft2: inverse col (complex) then weighted inverse-rfft row synthesis
  auto irfft2 = [&](const float* wr, const float* wi, float* sp, int batch) {
    gemm(Cf, wr, Hr, NPAD, NFP, NPAD, NPAD, NFP, NFP, 0, SPSZ, SPSZ,  1.f, 0.f, batch);
    gemm(Sf, wi, Hr, NPAD, NFP, NPAD, NPAD, NFP, NFP, 0, SPSZ, SPSZ, -1.f, 1.f, batch);
    gemm(Cf, wi, Hi, NPAD, NFP, NPAD, NPAD, NFP, NFP, 0, SPSZ, SPSZ,  1.f, 0.f, batch);
    gemm(Sf, wr, Hi, NPAD, NFP, NPAD, NPAD, NFP, NFP, 0, SPSZ, SPSZ,  1.f, 1.f, batch);
    gemm(Hr, IRr, sp, NPAD, NPAD, NFP, NFP, NPAD, NPAD, SPSZ, 0, IMSZ, 1.f, 0.f, batch);
    gemm(Hi, IRi, sp, NPAD, NPAD, NFP, NFP, NPAD, NPAD, SPSZ, 0, IMSZ, 1.f, 1.f, batch);
  };
  auto gnorm = [&](const float* x, float* y, const float* g, const float* b, int C, int gelu) {
    k_fill<<<1, 64, 0, stream>>>(stats, 8, 0.f);
    long long ns = (long long)C * HW;
    for (int s = 0; s < NB; ++s)
      k_gn_stats<<<512, 256, 0, stream>>>(x + s * ns, ns, stats + 2 * s);
    for (int s = 0; s < NB; ++s)
      k_gn_apply<<<GB(ns), 256, 0, stream>>>(x + s * ns, y + s * ns, g, b, stats + 2 * s, ns, C, gelu);
  };
  auto conv = [&](const float* x, const float* w, const float* b, float* y,
                  int Cin, int Cout, int kh) {
    int Kt = Cin * kh * kh;
    k_w2h<<<GB((long long)Cout * Kt), 256, 0, stream>>>(w, wh, Cout * Kt);
    dim3 g((NB * HW) / 32, Cout / 64, 1);
    k_conv_wmma<<<g, 256, 0, stream>>>(x, wh, b, y, Cin, Cout, kh);
  };

  const long long n64 = (long long)NB * CO * HW;
  auto rb = [&](const float* in, const RBP& P, float* out) {
    conv(in, P.conv1_w, P.conv1_b, xb, CI, CO, 3);
    gnorm(xb, xb, P.norm1_g, P.norm1_b, CO, 1);                      // x = gelu(gn(conv1))
    gnorm(xb, t64a, P.rn1_g, P.rn1_b, CO, 0);
    conv(t64a, P.pw1_w, P.pw1_b, t64b, CO, CO, 1);
    k_dw3<<<GB(n64), 256, 0, stream>>>(t64b, P.dw_w, P.dw_b, t64a);
    conv(t64a, P.pw2_w, P.pw2_b, t128, CO, 2 * CO, 1);
    k_gate<<<GB(n64), 256, 0, stream>>>(t128, t64b);
    k_chan_mean<<<NB * CO, 256, 0, stream>>>(xb, cm);
    k_se<<<1, 128, 0, stream>>>(cm, P.se_w, P.se_b, sev);
    k_scale_se<<<GB(n64), 256, 0, stream>>>(t64b, sev);
    conv(t64b, P.c3_w, P.c3_b, t64a, CO, CO, 3);
    k_axpy_chan<<<GB(n64), 256, 0, stream>>>(t64a, xb, P.beta, t64a, CO, n64);   // y = x + beta*y
    gnorm(t64a, t64b, P.rn2_g, P.rn2_b, CO, 0);
    conv(t64b, P.pw3_w, P.pw3_b, t128, CO, 2 * CO, 1);
    k_gate<<<GB(n64), 256, 0, stream>>>(t128, t64b);
    conv(t64b, P.pw4_w, P.pw4_b, t64c, CO, CO, 1);
    k_axpy_chan<<<GB(n64), 256, 0, stream>>>(t64a, t64a, P.gamma, t64c, CO, n64); // x = y + gamma*z
    conv(t64a, P.conv2_w, P.conv2_b, t64b, CO, CO, 3);
    gnorm(t64b, out, P.norm2_g, P.norm2_b, CO, 1);
  };

  // ================= pipeline =================
  k_init_dft<<<GB(IMSZ), 256, 0, stream>>>(Cn, Sn, Cf, Sf, IRr, IRi);

  // ---- ISO: inverse_cross = iso(forward_feature) ----
  k_pad<<<GB(64LL * IMSZ), 256, 0, stream>>>(fwdF, padA, 64, 1);     // edge pad
  k_gauss<<<GB(64LL * IMSZ), 256, 0, stream>>>(padA, padB, 64, 1);
  k_gauss<<<GB(64LL * IMSZ), 256, 0, stream>>>(padB, padA, 64, 0);
  rfft2(padA, Fr, Fi, Gr, Gi, 64);                                   // meas_fft
  k_red_sum<<<NK, 256, 0, stream>>>(psf, PHW * PHW, psfsum, 1.f);
  k_psf_pad_iso<<<GB((long long)NK * IMSZ), 256, 0, stream>>>(psf, psfsum, isoAlpha, psfPad);
  rfft2(psfPad, PFr, PFi, PGr, PGi, NK);                             // psf_fft (16)
  k_fill<<<GB((long long)NB * CI * HW), 256, 0, stream>>>(mixed, (long long)NB * CI * HW, 0.f);
  for (int k = 0; k < NK; ++k) {
    k_wiener<<<GB(64LL * SPSZ), 256, 0, stream>>>(Gr, Gi,
        PGr + (long long)k * SPSZ, PGi + (long long)k * SPSZ,
        isoPsfW + k * CI, Fr, Fi);                                   // reuse Fr/Fi
    irfft2(Fr, Fi, padB, 64);
    k_shift_crop<<<GB(64LL * HW), 256, 0, stream>>>(padB, roi + (long long)k * HW, mixed, 1);
  }
  gnorm(mixed, invX, isoGnG, isoGnB, CI, 0);

  // ---- FSO: forward_cross = fso(inverse_feature) ----
  k_pad<<<GB(64LL * IMSZ), 256, 0, stream>>>(invF, padA, 64, 0);     // zero pad
  rfft2(padA, Fr, Fi, Gr, Gi, 64);
  k_psf_mean<<<GB(PHW * PHW), 256, 0, stream>>>(psf, pm);
  k_red_sum<<<1, 256, 0, stream>>>(pm, PHW * PHW, ps, 1.f);
  k_psf_pad_fso<<<GB(IMSZ), 256, 0, stream>>>(pm, ps, fsoAlpha, psfPad);
  rfft2(psfPad, PFr, PFi, PGr, PGi, 1);
  k_cmul<<<GB(64LL * SPSZ), 256, 0, stream>>>(Gr, Gi, PGr, PGi, Fr, Fi);
  irfft2(Fr, Fi, padB, 64);
  k_shift_crop<<<GB(64LL * HW), 256, 0, stream>>>(padB, nullptr, mixed, 0);
  gnorm(mixed, fwdX, fsoGnG, fsoGnB, CI, 0);

  // ---- mix + residual blocks ----
  long long n32 = (long long)NB * CI * HW;
  k_mix<<<GB(n32), 256, 0, stream>>>(inA, invF, invMixS, invX, invMixC, n32);
  k_mix<<<GB(n32), 256, 0, stream>>>(inB, fwdF, fwdMixS, fwdX, fwdMixC, n32);

  float* dout = (float*)d_out;
  rb(inA, rbI, dout);
  rb(inB, rbF, dout + (size_t)NB * CO * HW);
}
